// MedicalGNNEncoder_18640158064961
// MI455X (gfx1250) — compile-verified, matrix operand order checked by reading, not run on hardware
//
#include <hip/hip_runtime.h>
#include <math.h>

// ---------------------------------------------------------------------------
// Types for CDNA5 WMMA
// ---------------------------------------------------------------------------
typedef __attribute__((ext_vector_type(16))) __bf16 v16bf;
typedef __attribute__((ext_vector_type(8)))  float  v8f;

#define NNODES 50000
#define NEDGES 800000
#define NGRAPHS 64
#define CDIM 128

static __device__ __forceinline__ unsigned short f32_to_bf16_bits(float f) {
    unsigned u = __float_as_uint(f);
    u += 0x7FFFu + ((u >> 16) & 1u);   // round-to-nearest-even
    return (unsigned short)(u >> 16);
}

static __device__ __forceinline__ void atomicMaxF(float* addr, float val) {
    if (val >= 0.0f) atomicMax((int*)addr, __float_as_int(val));
    else             atomicMin((unsigned int*)addr, __float_as_uint(val));
}

// ---------------------------------------------------------------------------
// Generic fill
// ---------------------------------------------------------------------------
__global__ void k_fill_f32(float* __restrict__ p, size_t n, float v) {
    size_t i = (size_t)blockIdx.x * blockDim.x + threadIdx.x;
    if (i < n) p[i] = v;
}

// ---------------------------------------------------------------------------
// Weight prep: W[din][F] f32  ->  Wt[F][Kpad] bf16 (zero padded K)
// ---------------------------------------------------------------------------
__global__ void k_prep_weight(const float* __restrict__ W,
                              unsigned short* __restrict__ Wt,
                              int din, int F, int Kpad) {
    int idx = blockIdx.x * blockDim.x + threadIdx.x;
    int total = F * Kpad;
    if (idx >= total) return;
    int j = idx / Kpad;
    int k = idx - j * Kpad;
    float v = (k < din) ? W[(size_t)k * F + j] : 0.0f;
    Wt[idx] = f32_to_bf16_bits(v);
}

// x[N][8] f32 -> Xb[N][32] bf16 (zero padded)
__global__ void k_prep_x(const float* __restrict__ x,
                         unsigned short* __restrict__ Xb, int N) {
    int idx = blockIdx.x * blockDim.x + threadIdx.x;
    if (idx >= N * 32) return;
    int n = idx >> 5, k = idx & 31;
    Xb[idx] = (k < 8) ? f32_to_bf16_bits(x[(size_t)n * 8 + k]) : (unsigned short)0;
}

// ---------------------------------------------------------------------------
// Self-loop attr = mean of incoming edge attrs
// ---------------------------------------------------------------------------
__global__ void k_loop_accum(const int* __restrict__ dst,
                             const float* __restrict__ edge_attr,
                             float* __restrict__ cnt, float* __restrict__ asum, int E) {
    int e = blockIdx.x * blockDim.x + threadIdx.x;
    if (e >= E) return;
    int d = dst[e];
    atomicAdd(&cnt[d], 1.0f);
    atomicAdd(&asum[(size_t)d * 3 + 0], edge_attr[(size_t)e * 3 + 0]);
    atomicAdd(&asum[(size_t)d * 3 + 1], edge_attr[(size_t)e * 3 + 1]);
    atomicAdd(&asum[(size_t)d * 3 + 2], edge_attr[(size_t)e * 3 + 2]);
}

__global__ void k_loop_final(const float* __restrict__ cnt, const float* __restrict__ asum,
                             float* __restrict__ lattr, int N) {
    int n = blockIdx.x * blockDim.x + threadIdx.x;
    if (n >= N) return;
    float c = cnt[n]; c = c > 1.0f ? c : 1.0f;
    lattr[(size_t)n * 3 + 0] = asum[(size_t)n * 3 + 0] / c;
    lattr[(size_t)n * 3 + 1] = asum[(size_t)n * 3 + 1] / c;
    lattr[(size_t)n * 3 + 2] = asum[(size_t)n * 3 + 2] / c;
}

// ---------------------------------------------------------------------------
// GEMM: Y[N][F] = Xb[N][Kpad](bf16) @ Wt[F][Kpad]^T(bf16) + bias
// One wave per 16x64 output strip: per K-step load 1 A frag + 4 B frags,
// issue 4 independent v_wmma_f32_16x16x32_bf16 (A reuse, 4 acc chains).
// ---------------------------------------------------------------------------
__global__ void k_gemm_bf16_wmma(const unsigned short* __restrict__ Xb,
                                 const unsigned short* __restrict__ Wt,
                                 const float* __restrict__ bias,
                                 float* __restrict__ Y,
                                 int N, int Kpad, int F) {
    int lane = threadIdx.x & 31;
    int wave = threadIdx.x >> 5;
    int rowTile = blockIdx.x * (blockDim.x >> 5) + wave;
    int RT = N >> 4;
    if (rowTile >= RT) return;
    int m0 = rowTile << 4;
    int n0 = blockIdx.y << 6;                 // 64 output columns per wave
    int lo = lane & 15, hi = lane >> 4;

    const unsigned short* arow = Xb + (size_t)(m0 + lo) * Kpad + 8 * hi;
    const unsigned short* brow = Wt + (size_t)(n0 + lo) * Kpad + 8 * hi;
    size_t bstride = (size_t)16 * Kpad;       // next 16-column tile of Wt

    v8f acc0 = {0.f,0.f,0.f,0.f,0.f,0.f,0.f,0.f};
    v8f acc1 = acc0, acc2 = acc0, acc3 = acc0;

    for (int k0 = 0; k0 < Kpad; k0 += 32) {
        union { v16bf v; uint4 u[2]; } A, B0, B1, B2, B3;
        A.u[0]  = *(const uint4*)(arow + k0);
        A.u[1]  = *(const uint4*)(arow + k0 + 16);
        B0.u[0] = *(const uint4*)(brow + k0);
        B0.u[1] = *(const uint4*)(brow + k0 + 16);
        B1.u[0] = *(const uint4*)(brow + bstride + k0);
        B1.u[1] = *(const uint4*)(brow + bstride + k0 + 16);
        B2.u[0] = *(const uint4*)(brow + 2 * bstride + k0);
        B2.u[1] = *(const uint4*)(brow + 2 * bstride + k0 + 16);
        B3.u[0] = *(const uint4*)(brow + 3 * bstride + k0);
        B3.u[1] = *(const uint4*)(brow + 3 * bstride + k0 + 16);
        acc0 = __builtin_amdgcn_wmma_f32_16x16x32_bf16(false, A.v, false, B0.v,
                                                       (short)0, acc0, false, false);
        acc1 = __builtin_amdgcn_wmma_f32_16x16x32_bf16(false, A.v, false, B1.v,
                                                       (short)0, acc1, false, false);
        acc2 = __builtin_amdgcn_wmma_f32_16x16x32_bf16(false, A.v, false, B2.v,
                                                       (short)0, acc2, false, false);
        acc3 = __builtin_amdgcn_wmma_f32_16x16x32_bf16(false, A.v, false, B3.v,
                                                       (short)0, acc3, false, false);
    }

    float bv0 = bias[n0 + lo];
    float bv1 = bias[n0 + 16 + lo];
    float bv2 = bias[n0 + 32 + lo];
    float bv3 = bias[n0 + 48 + lo];
#pragma unroll
    for (int r = 0; r < 8; ++r) {
        float* yr = Y + (size_t)(m0 + r + 8 * hi) * F + n0 + lo;
        yr[0]  = acc0[r] + bv0;
        yr[16] = acc1[r] + bv1;
        yr[32] = acc2[r] + bv2;
        yr[48] = acc3[r] + bv3;
    }
}

// ---------------------------------------------------------------------------
// Edge pass 1: logit[e][h] = sum_c att[h][c]*leakyrelu(xl[s]+xr[d]+ea@We, 0.2)
// One wave per edge (self-loops implicit for e >= E). atomicMax into Mmax.
// ---------------------------------------------------------------------------
__global__ void k_edge_logit(const int* __restrict__ src, const int* __restrict__ dst,
                             const float* __restrict__ edge_attr,
                             const float* __restrict__ lattr,
                             const float* __restrict__ XL, const float* __restrict__ XR,
                             const float* __restrict__ We, const float* __restrict__ att,
                             float* __restrict__ ALOG, float* __restrict__ Mmax,
                             int E, int N, int H, int F) {
    int lane = threadIdx.x & 31;
    int e = blockIdx.x * (blockDim.x >> 5) + (threadIdx.x >> 5);
    int E2 = E + N;
    if (e >= E2) return;
    int s, d; const float* ea;
    if (e < E) { s = src[e]; d = dst[e]; ea = edge_attr + (size_t)e * 3; }
    else       { s = d = e - E;          ea = lattr     + (size_t)(e - E) * 3; }
    float ea0 = ea[0], ea1 = ea[1], ea2 = ea[2];
    int V = F >> 5;
    int j0 = lane * V;
    const float* xlp = XL + (size_t)s * F + j0;
    const float* xrp = XR + (size_t)d * F + j0;
    float acc = 0.0f;
    for (int v = 0; v < V; v += 4) {
        float4 a4 = *(const float4*)(xlp + v);
        float4 b4 = *(const float4*)(xrp + v);
#pragma unroll
        for (int q = 0; q < 4; ++q) {
            int j = j0 + v + q;
            float ee = ea0 * We[j] + ea1 * We[F + j] + ea2 * We[2 * F + j];
            float sv = (&a4.x)[q] + (&b4.x)[q] + ee;
            sv = sv > 0.0f ? sv : 0.2f * sv;
            acc += sv * att[j];
        }
    }
    int lph = 32 / H;  // lanes per head (contiguous)
    for (int off = lph >> 1; off; off >>= 1) acc += __shfl_down(acc, off, 32);
    if ((lane & (lph - 1)) == 0) {
        int h = lane / lph;
        ALOG[(size_t)e * H + h] = acc;
        atomicMaxF(&Mmax[(size_t)d * H + h], acc);
    }
}

// Edge pass 2: a = exp(logit - m[dst]); denom[dst] += a
__global__ void k_edge_softmax(const int* __restrict__ dst,
                               float* __restrict__ ALOG,
                               const float* __restrict__ Mmax,
                               float* __restrict__ DEN,
                               int E, int N, int H) {
    int idx = blockIdx.x * blockDim.x + threadIdx.x;
    int total = (E + N) * H;
    if (idx >= total) return;
    int e = idx / H, h = idx - e * H;
    int d = (e < E) ? dst[e] : (e - E);
    float a = __expf(ALOG[idx] - Mmax[(size_t)d * H + h]);
    ALOG[idx] = a;
    atomicAdd(&DEN[(size_t)d * H + h], a);
}

// Edge pass 3: OUT[dst] += xl[src] * a/(den+eps)  (OUT pre-init to bo)
__global__ void k_edge_aggregate(const int* __restrict__ src, const int* __restrict__ dst,
                                 const float* __restrict__ XL,
                                 const float* __restrict__ ALOG, const float* __restrict__ DEN,
                                 float* __restrict__ OUT,
                                 int E, int N, int H, int F) {
    int lane = threadIdx.x & 31;
    int e = blockIdx.x * (blockDim.x >> 5) + (threadIdx.x >> 5);
    int E2 = E + N;
    if (e >= E2) return;
    int s, d;
    if (e < E) { s = src[e]; d = dst[e]; } else { s = d = e - E; }
    int V = F >> 5;
    int j0 = lane * V;
    int h = j0 >> 7;  // /C
    float w = ALOG[(size_t)e * H + h] / (DEN[(size_t)d * H + h] + 1e-16f);
    const float* xlp = XL + (size_t)s * F + j0;
    float* op = OUT + (size_t)d * F + j0;
    for (int v = 0; v < V; ++v) atomicAdd(op + v, xlp[v] * w);
}

// OUT[n][j] = bo[j]
__global__ void k_init_out(float* __restrict__ OUT, const float* __restrict__ bo,
                           size_t total, int Fmask) {
    size_t i = (size_t)blockIdx.x * blockDim.x + threadIdx.x;
    if (i < total) OUT[i] = bo[i & (size_t)Fmask];
}

// ---------------------------------------------------------------------------
// Graph layernorm stats: S[0] = sum, S[1] = sum of squares (f64 atomics)
// ---------------------------------------------------------------------------
__global__ void k_reduce_stats(const float* __restrict__ X, size_t M, double* __restrict__ S) {
    __shared__ double sa[256], sb[256];
    double a = 0.0, b = 0.0;
    for (size_t i = (size_t)blockIdx.x * blockDim.x + threadIdx.x; i < M;
         i += (size_t)gridDim.x * blockDim.x) {
        double v = (double)X[i];
        a += v; b += v * v;
    }
    sa[threadIdx.x] = a; sb[threadIdx.x] = b;
    __syncthreads();
    for (int st = blockDim.x >> 1; st; st >>= 1) {
        if ((int)threadIdx.x < st) {
            sa[threadIdx.x] += sa[threadIdx.x + st];
            sb[threadIdx.x] += sb[threadIdx.x + st];
        }
        __syncthreads();
    }
    if (threadIdx.x == 0) { atomicAdd(&S[0], sa[0]); atomicAdd(&S[1], sb[0]); }
}

// y = elu((x-mu)/(sd+1e-5)*w + b); write f32 (in place) and bf16 for next GEMM
__global__ void k_ln_elu_apply(float* __restrict__ OUT,
                               const float* __restrict__ lnw, const float* __restrict__ lnb,
                               const double* __restrict__ S,
                               unsigned short* __restrict__ Xb, int writeXb,
                               size_t total, int Fmask) {
    size_t i = (size_t)blockIdx.x * blockDim.x + threadIdx.x;
    if (i >= total) return;
    double invM = 1.0 / (double)total;
    double mu = S[0] * invM;
    double var = S[1] * invM - mu * mu;
    float sd = (float)sqrt(var > 0.0 ? var : 0.0);
    int j = (int)(i & (size_t)Fmask);
    float v = OUT[i];
    float y = (v - (float)mu) / (sd + 1e-5f) * lnw[j] + lnb[j];
    y = y > 0.0f ? y : (__expf(y) - 1.0f);
    OUT[i] = y;
    if (writeXb) Xb[i] = f32_to_bf16_bits(y);
}

// ---------------------------------------------------------------------------
// Pooling: mean || max per graph over nodes (F = 128)
// ---------------------------------------------------------------------------
__global__ void k_pool_accum(const float* __restrict__ h, const int* __restrict__ batch,
                             float* __restrict__ meanws, float* __restrict__ maxws,
                             float* __restrict__ gcnt, int N) {
    int idx = blockIdx.x * blockDim.x + threadIdx.x;
    if (idx >= N * CDIM) return;
    int n = idx >> 7, j = idx & 127;
    int g = batch[n];
    float v = h[idx];
    atomicAdd(&meanws[g * CDIM + j], v);
    atomicMaxF(&maxws[g * CDIM + j], v);
    if (j == 0) atomicAdd(&gcnt[g], 1.0f);
}

__global__ void k_pool_final(const float* __restrict__ meanws, const float* __restrict__ maxws,
                             const float* __restrict__ gcnt, float* __restrict__ out) {
    int idx = blockIdx.x * blockDim.x + threadIdx.x;
    if (idx >= NGRAPHS * CDIM) return;
    int g = idx >> 7, j = idx & 127;
    float c = gcnt[g]; c = c > 1.0f ? c : 1.0f;
    out[(size_t)g * 256 + j] = meanws[idx] / c;
    float mx = maxws[idx];
    out[(size_t)g * 256 + 128 + j] = isfinite(mx) ? mx : 0.0f;
}

// ---------------------------------------------------------------------------
// Host driver
// ---------------------------------------------------------------------------
static inline int ceil_div(long long a, long long b) { return (int)((a + b - 1) / b); }

extern "C" void kernel_launch(void* const* d_in, const int* in_sizes, int n_in,
                              void* d_out, int out_size, void* d_ws, size_t ws_size,
                              hipStream_t stream) {
    const int N = NNODES, E = NEDGES, E2 = NEDGES + NNODES;

    const float* x         = (const float*)d_in[0];
    const int*   src       = (const int*)d_in[1];
    const int*   dst       = ((const int*)d_in[1]) + E;
    const float* edge_attr = (const float*)d_in[2];
    const int*   batch     = (const int*)d_in[3];

    // layer configs
    const int din[3]  = {8, 512, 256};
    const int Hh[3]   = {4, 2, 1};
    const int Kpad[3] = {32, 512, 256};
    const int Fo[3]   = {512, 256, 128};

    const float *Wl[3], *bl[3], *Wr[3], *br[3], *We[3], *att[3], *bo[3], *lnw[3], *lnb[3];
    for (int l = 0; l < 3; ++l) {
        int base = 4 + 9 * l;
        Wl[l]  = (const float*)d_in[base + 0];
        bl[l]  = (const float*)d_in[base + 1];
        Wr[l]  = (const float*)d_in[base + 2];
        br[l]  = (const float*)d_in[base + 3];
        We[l]  = (const float*)d_in[base + 4];
        att[l] = (const float*)d_in[base + 5];
        bo[l]  = (const float*)d_in[base + 6];
        lnw[l] = (const float*)d_in[base + 7];
        lnb[l] = (const float*)d_in[base + 8];
    }

    // ---- workspace carve (256B aligned) ----
    size_t off = 0;
    char* base = (char*)d_ws;
    auto carve = [&](size_t bytes) -> char* {
        char* p = base + off;
        off += (bytes + 255) & ~(size_t)255;
        return p;
    };
    unsigned short* Xb  = (unsigned short*)carve((size_t)N * 512 * 2);
    float* XL   = (float*)carve((size_t)N * 512 * 4);
    float* XR   = (float*)carve((size_t)N * 512 * 4);
    float* OUT  = (float*)carve((size_t)N * 512 * 4);
    float* ALOG = (float*)carve((size_t)E2 * 4 * 4);
    float* Mmax = (float*)carve((size_t)N * 4 * 4);
    float* DEN  = (float*)carve((size_t)N * 4 * 4);
    float* cnt  = (float*)carve((size_t)N * 4);
    float* asum = (float*)carve((size_t)N * 3 * 4);
    float* lattr= (float*)carve((size_t)N * 3 * 4);
    unsigned short* Wlt[3]; unsigned short* Wrt[3];
    for (int l = 0; l < 3; ++l) {
        Wlt[l] = (unsigned short*)carve((size_t)Fo[l] * Kpad[l] * 2);
        Wrt[l] = (unsigned short*)carve((size_t)Fo[l] * Kpad[l] * 2);
    }
    double* S      = (double*)carve(16);
    float* meanws  = (float*)carve((size_t)NGRAPHS * CDIM * 4);
    float* maxws   = (float*)carve((size_t)NGRAPHS * CDIM * 4);
    float* gcnt    = (float*)carve((size_t)NGRAPHS * 4);
    (void)ws_size; (void)n_in; (void)in_sizes; (void)out_size;

    const int TB = 256;

    // ---- weight prep (transpose + bf16) ----
    for (int l = 0; l < 3; ++l) {
        int tot = Fo[l] * Kpad[l];
        k_prep_weight<<<ceil_div(tot, TB), TB, 0, stream>>>(Wl[l], Wlt[l], din[l], Fo[l], Kpad[l]);
        k_prep_weight<<<ceil_div(tot, TB), TB, 0, stream>>>(Wr[l], Wrt[l], din[l], Fo[l], Kpad[l]);
    }
    // ---- input prep ----
    k_prep_x<<<ceil_div((long long)N * 32, TB), TB, 0, stream>>>(x, Xb, N);

    // ---- self-loop attrs ----
    k_fill_f32<<<ceil_div(N, TB), TB, 0, stream>>>(cnt, (size_t)N, 0.0f);
    k_fill_f32<<<ceil_div((long long)N * 3, TB), TB, 0, stream>>>(asum, (size_t)N * 3, 0.0f);
    k_loop_accum<<<ceil_div(E, TB), TB, 0, stream>>>(dst, edge_attr, cnt, asum, E);
    k_loop_final<<<ceil_div(N, TB), TB, 0, stream>>>(cnt, asum, lattr, N);

    // ---- layers ----
    for (int l = 0; l < 3; ++l) {
        int H = Hh[l], F = Fo[l], K = Kpad[l];
        size_t totNF = (size_t)N * F;

        // GEMMs: xl, xr  (WMMA bf16, 16x64 strip per wave)
        dim3 ggrid(ceil_div(N / 16, 8), F / 64);
        k_gemm_bf16_wmma<<<ggrid, 256, 0, stream>>>(Xb, Wlt[l], bl[l], XL, N, K, F);
        k_gemm_bf16_wmma<<<ggrid, 256, 0, stream>>>(Xb, Wrt[l], br[l], XR, N, K, F);

        // softmax state init
        k_fill_f32<<<ceil_div((long long)N * H, TB), TB, 0, stream>>>(Mmax, (size_t)N * H, -INFINITY);
        k_fill_f32<<<ceil_div((long long)N * H, TB), TB, 0, stream>>>(DEN, (size_t)N * H, 0.0f);

        // edge passes
        int eblocks = ceil_div(E2, 8);  // 8 waves per 256-thread block
        k_edge_logit<<<eblocks, 256, 0, stream>>>(src, dst, edge_attr, lattr, XL, XR,
                                                  We[l], att[l], ALOG, Mmax, E, N, H, F);
        k_edge_softmax<<<ceil_div((long long)E2 * H, TB), TB, 0, stream>>>(dst, ALOG, Mmax, DEN, E, N, H);
        k_init_out<<<ceil_div((long long)totNF, TB), TB, 0, stream>>>(OUT, bo[l], totNF, F - 1);
        k_edge_aggregate<<<eblocks, 256, 0, stream>>>(src, dst, XL, ALOG, DEN, OUT, E, N, H, F);

        // graph layernorm + ELU (+ bf16 for next layer's GEMM)
        k_fill_f32<<<1, 4, 0, stream>>>((float*)S, 4, 0.0f);  // zero two doubles
        k_reduce_stats<<<2048, 256, 0, stream>>>(OUT, totNF, S);
        k_ln_elu_apply<<<ceil_div((long long)totNF, TB), TB, 0, stream>>>(
            OUT, lnw[l], lnb[l], S, Xb, (l < 2) ? 1 : 0, totNF, F - 1);
    }

    // ---- pooling: mean || max over graphs ----
    k_fill_f32<<<ceil_div(NGRAPHS * CDIM, TB), TB, 0, stream>>>(meanws, (size_t)NGRAPHS * CDIM, 0.0f);
    k_fill_f32<<<ceil_div(NGRAPHS * CDIM, TB), TB, 0, stream>>>(maxws, (size_t)NGRAPHS * CDIM, -INFINITY);
    k_fill_f32<<<1, NGRAPHS, 0, stream>>>(gcnt, (size_t)NGRAPHS, 0.0f);
    k_pool_accum<<<ceil_div((long long)N * CDIM, TB), TB, 0, stream>>>(OUT, batch, meanws, maxws, gcnt, N);
    k_pool_final<<<ceil_div(NGRAPHS * CDIM, TB), TB, 0, stream>>>(meanws, maxws, gcnt, (float*)d_out);
}
